// CausalAttention_29042568855934
// MI455X (gfx1250) — matrix operand order repaired and checked by reading, not compiled, not run on hardware
//
#include <hip/hip_runtime.h>
#include <math.h>

// ---------------------------------------------------------------------------
// CausalAttention on gfx1250 (MI455X): bf16 WMMA, double-buffered async-to-LDS
// staging (ASYNCcnt pipelining).   B=1, S=4096, D=1024, H=16, HD=64
// ---------------------------------------------------------------------------

typedef __attribute__((ext_vector_type(16))) __bf16 v16bf;
typedef __attribute__((ext_vector_type(8)))  float  v8f;

#define S_LEN   4096
#define D_MODEL 1024
#define NHEADS  16
#define HDIM    64

// ---- helpers ---------------------------------------------------------------

static __device__ __forceinline__ unsigned short f32_to_bf16(float f) {
    unsigned int u = __float_as_uint(f);
    u += 0x7FFFu + ((u >> 16) & 1u);   // round-to-nearest-even
    return (unsigned short)(u >> 16);
}

union FragBF16 { v16bf v; unsigned short s[16]; };

static __device__ __forceinline__ v8f wmma_bf16(v16bf a, v16bf b, v8f c) {
    return __builtin_amdgcn_wmma_f32_16x16x32_bf16(
        false, a, false, b, (short)0, c, false, false);
}

// A-matrix 16x32 bf16 fragment from row-major [rows][ldm] (LDS or global).
// m=lane&15, half=lane>>4; elems 0-15: K=koff+half*16+e (one 32B run).
static __device__ __forceinline__ v16bf load_a_frag(const unsigned short* mem,
                                                    int ldm, int row_base,
                                                    int koff, int lane) {
    const int m = lane & 15, half = lane >> 4;
    const unsigned short* p = mem + (size_t)(row_base + m) * ldm + koff;
    FragBF16 u;
#pragma unroll
    for (int e = 0; e < 16; ++e) u.s[e] = p[half * 16 + e];
    return u.v;
}

// B-matrix 32x16 bf16 fragment from K-contiguous storage mem[n][k], stride ldk.
// Lane: n=lane&15, kh=lane>>4; elems 0-15: K=koff+kh*16+e (one 32B run).
static __device__ __forceinline__ v16bf load_b_frag_t(const unsigned short* mem,
                                                      int ldk, int row_base,
                                                      int koff, int lane) {
    const int n = lane & 15, kh = lane >> 4;
    const unsigned short* p = mem + (size_t)(row_base + n) * ldk + koff + kh * 16;
    FragBF16 u;
#pragma unroll
    for (int e = 0; e < 16; ++e) u.s[e] = p[e];
    return u.v;
}

static __device__ __forceinline__ v8f zero_v8f() {
    v8f z;
#pragma unroll
    for (int i = 0; i < 8; ++i) z[i] = 0.0f;
    return z;
}

// ---- CDNA5 async-to-LDS staging (CDNA5 ISA) ----------
static __device__ __forceinline__ void async_load_b128(void* lds, const void* g) {
    unsigned           l  = (unsigned)(size_t)lds;           // LDS offset
    unsigned long long ga = (unsigned long long)(size_t)g;   // global addr
    asm volatile("global_load_async_to_lds_b128 %0, %1, off"
                 :: "v"(l), "v"(ga) : "memory");
}
// Asymmetric wait: async loads complete in order, so waiting ASYNCcnt<=N
// drains everything except the N most recently issued ops.
template <int N>
static __device__ __forceinline__ void wait_async_le() {
    asm volatile("s_wait_asynccnt %0" :: "i"(N) : "memory");
}

// ---- kernel 1a: fp32 -> bf16 elementwise (for x) ---------------------------

__global__ void cvt_f32_bf16_kernel(const float* __restrict__ in,
                                    unsigned short* __restrict__ out, int n) {
    int i = (blockIdx.x * blockDim.x + threadIdx.x) * 4;
#pragma unroll
    for (int j = 0; j < 4; ++j)
        if (i + j < n) out[i + j] = f32_to_bf16(in[i + j]);
}

// ---- kernel 1b: fp32 [1024][N] -> bf16 transposed [N][1024] ---------------

__global__ void cvt_transpose_kernel(const float* __restrict__ in,
                                     unsigned short* __restrict__ out, int N) {
    int idx = blockIdx.x * blockDim.x + threadIdx.x;   // linear over [N][1024]
    int n = idx >> 10, k = idx & 1023;                 // K = D_MODEL = 1024
    if (n < N) out[idx] = f32_to_bf16(in[(size_t)k * N + n]);
}

// ---- GEMM core: 64(M) x 128(N) workgroup tile, K-step 32, double-buffered --
// block 256 = 8 waves (2x4), wave does 32x32 via 4 WMMA per K-step.
// A staged row-major [m][k]; B staged K-contiguous [n][k] from transposed W.
// Per-tile async ops per thread: 1 (A) + 2 (B) = 3.

static __device__ __forceinline__ void
gemm_stage(const unsigned short* A, const unsigned short* BT,
           unsigned short* As, unsigned short* Bs,
           int bm, int bn, int k0, int tid) {
    {
        int idx = tid * 8;                       // A: 64x32, 1 chunk/thread
        int m = idx >> 5, kk = idx & 31;
        async_load_b128(&As[idx], A + (size_t)(bm * 64 + m) * D_MODEL + k0 + kk);
    }
#pragma unroll
    for (int it = 0; it < 2; ++it) {             // B: 128x32, 2 chunks/thread
        int idx = tid * 8 + it * 2048;
        int n = idx >> 5, kk = idx & 31;
        async_load_b128(&Bs[idx], BT + (size_t)(bn * 128 + n) * D_MODEL + k0 + kk);
    }
}

struct GemmAcc { v8f c00, c01, c10, c11; };

static __device__ __forceinline__ GemmAcc
gemm_mainloop(const unsigned short* A, const unsigned short* BT,
              unsigned short* As, unsigned short* Bs,   // [2][tile]
              int bm, int bn, int tid, int lane, int wm, int wn) {
    GemmAcc g;
    g.c00 = zero_v8f(); g.c01 = zero_v8f(); g.c10 = zero_v8f(); g.c11 = zero_v8f();

    gemm_stage(A, BT, As, Bs, bm, bn, 0, tid);                 // prologue
    const int NT = D_MODEL / 32;
    for (int t = 0; t < NT; ++t) {
        unsigned short* Asb = As + (t & 1) * (64 * 32);
        unsigned short* Bsb = Bs + (t & 1) * (128 * 32);
        if (t + 1 < NT) {
            gemm_stage(A, BT, As + ((t + 1) & 1) * (64 * 32),
                       Bs + ((t + 1) & 1) * (128 * 32), bm, bn,
                       (t + 1) * 32, tid);
            wait_async_le<3>();    // drain tile t, keep tile t+1 in flight
        } else {
            wait_async_le<0>();
        }
        __syncthreads();           // tile t visible to all waves

        v16bf a0 = load_a_frag(Asb, 32, wm * 32,      0, lane);
        v16bf a1 = load_a_frag(Asb, 32, wm * 32 + 16, 0, lane);
        v16bf b0 = load_b_frag_t(Bsb, 32, wn * 32,      0, lane);
        v16bf b1 = load_b_frag_t(Bsb, 32, wn * 32 + 16, 0, lane);
        g.c00 = wmma_bf16(a0, b0, g.c00);
        g.c01 = wmma_bf16(a0, b1, g.c01);
        g.c10 = wmma_bf16(a1, b0, g.c10);
        g.c11 = wmma_bf16(a1, b1, g.c11);
        __syncthreads();           // all reads of this buffer done
    }
    return g;
}

// ---- kernel 2: QKV GEMM + bias -> Q(scaled),K [h][s][hd], V [h][hd][s] -----

__global__ void __launch_bounds__(256)
qkv_gemm_kernel(const unsigned short* __restrict__ xb,      // [4096][1024]
                const unsigned short* __restrict__ wT,      // [3072][1024]
                const float* __restrict__ bias,
                unsigned short* __restrict__ Qb,
                unsigned short* __restrict__ Kb,
                unsigned short* __restrict__ Vt) {
    __shared__ unsigned short As[2 * 64 * 32];
    __shared__ unsigned short Bs[2 * 128 * 32];

    const int tid = threadIdx.x, lane = tid & 31, wave = tid >> 5;
    const int wm = wave >> 2, wn = wave & 3;
    const int bm = blockIdx.y, bn = blockIdx.x;

    GemmAcc g = gemm_mainloop(xb, wT, As, Bs, bm, bn, tid, lane, wm, wn);

    const int half = lane >> 4, nlane = lane & 15;
    v8f acc[2][2] = {{g.c00, g.c01}, {g.c10, g.c11}};
#pragma unroll
    for (int ti = 0; ti < 2; ++ti)
#pragma unroll
        for (int tj = 0; tj < 2; ++tj)
#pragma unroll
            for (int r = 0; r < 8; ++r) {
                int row = bm * 64 + wm * 32 + ti * 16 + half * 8 + r;
                int col = bn * 128 + wn * 32 + tj * 16 + nlane;    // 0..3071
                float val = acc[ti][tj][r] + bias[col];
                int which = col >> 10;                 // 0=q 1=k 2=v
                int d  = col & (D_MODEL - 1);
                int hh = d >> 6, hd = d & (HDIM - 1);
                if (which == 0)
                    Qb[((size_t)hh * S_LEN + row) * HDIM + hd] =
                        f32_to_bf16(val * 0.125f);     // fold 1/sqrt(64)
                else if (which == 1)
                    Kb[((size_t)hh * S_LEN + row) * HDIM + hd] = f32_to_bf16(val);
                else
                    Vt[((size_t)hh * HDIM + hd) * S_LEN + row] = f32_to_bf16(val);
            }
}

// ---- kernel 3: causal flash attention --------------------------------------
// grid (S/128, H), block 256 = 8 waves; wave w owns rows qblk*128+w*16..+15.
// Ks: [key][hd] (Q@K^T B-frag contiguous in hd), Vs: [hd][key] (P@V B-frag
// contiguous in key). K/V double-buffered; 4 async ops/thread/tile.

__global__ void __launch_bounds__(256)
flash_attn_kernel(const unsigned short* __restrict__ Qb,   // [h][s][hd]
                  const unsigned short* __restrict__ Kb,   // [h][s][hd]
                  const unsigned short* __restrict__ Vt,   // [h][hd][s]
                  unsigned short* __restrict__ Ob) {       // [s][d]
    __shared__ unsigned short Ks[2 * 64 * 64];
    __shared__ unsigned short Vs[2 * 64 * 64];
    __shared__ unsigned short Ps[8 * 16 * 64];   // per-wave P staging

    const int tid  = threadIdx.x;
    const int lane = tid & 31;
    const int wave = tid >> 5;
    const int half = lane >> 4, nlane = lane & 15;
    const int qblk = blockIdx.x;         // block of 128 query rows
    const int h    = blockIdx.y;
    const int qs_w = qblk * 128 + wave * 16;
    const int kb_max = 2 * qblk + 1;     // inclusive: keys up to qblk*128+127

    const unsigned short* kgbase = Kb + (size_t)h * S_LEN * HDIM;
    const unsigned short* vgbase = Vt + (size_t)h * HDIM * S_LEN;

    // Q fragments (16x64 -> two 16x32 A-frags), loaded once from global
    const unsigned short* qbase = Qb + (size_t)h * S_LEN * HDIM;
    v16bf qf[2];
#pragma unroll
    for (int kk = 0; kk < 2; ++kk)
        qf[kk] = load_a_frag(qbase, HDIM, qs_w, kk * 32, lane);

    v8f o[4];
#pragma unroll
    for (int t = 0; t < 4; ++t) o[t] = zero_v8f();
    float m_run[8], l_run[8];
#pragma unroll
    for (int r = 0; r < 8; ++r) { m_run[r] = -1e30f; l_run[r] = 0.0f; }

    unsigned short* Pw = &Ps[wave * 16 * 64];

    // stage K [key][hd] + V [hd][key] for key block kb: 512 chunks each,
    // 2 chunks per thread per matrix (256 threads)
    auto stage_kv = [&](int kb) {
        unsigned short* Kd = Ks + (kb & 1) * (64 * 64);
        unsigned short* Vd = Vs + (kb & 1) * (64 * 64);
        const int ks = kb * 64;
#pragma unroll
        for (int i = 0; i < 2; ++i) {
            int idx = (tid + i * 256) * 8;          // 0..4095
            int r0 = idx >> 6, c0 = idx & 63;
            async_load_b128(&Kd[idx], kgbase + (size_t)(ks + r0) * HDIM + c0);
            async_load_b128(&Vd[idx], vgbase + (size_t)r0 * S_LEN + ks + c0);
        }
    };

    stage_kv(0);                                     // prologue
    for (int kb = 0; kb <= kb_max; ++kb) {
        const int ks = kb * 64;
        const unsigned short* Ksb = Ks + (kb & 1) * (64 * 64);
        const unsigned short* Vsb = Vs + (kb & 1) * (64 * 64);
        if (kb + 1 <= kb_max) { stage_kv(kb + 1); wait_async_le<4>(); }
        else                  { wait_async_le<0>(); }
        __syncthreads();

        // S = Q @ K^T : 4 tiles of 16x16 over the 64 keys
        v8f st[4];
#pragma unroll
        for (int t = 0; t < 4; ++t) {
            v8f s = zero_v8f();
#pragma unroll
            for (int kk = 0; kk < 2; ++kk)
                s = wmma_bf16(qf[kk],
                              load_b_frag_t(Ksb, HDIM, t * 16, kk * 32, lane), s);
            st[t] = s;
        }

        // causal mask + online softmax (row = half*8+r, col = t*16+nlane)
        float alpha[8];
#pragma unroll
        for (int r = 0; r < 8; ++r) {
            int row = qs_w + half * 8 + r;
#pragma unroll
            for (int t = 0; t < 4; ++t) {
                int col = ks + t * 16 + nlane;
                if (col > row) st[t][r] = -1e30f;
            }
            float mx = st[0][r];
#pragma unroll
            for (int t = 1; t < 4; ++t) mx = fmaxf(mx, st[t][r]);
#pragma unroll
            for (int msk = 1; msk < 16; msk <<= 1)
                mx = fmaxf(mx, __shfl_xor(mx, msk, 32));
            float m_new = fmaxf(m_run[r], mx);
            alpha[r] = __expf(m_run[r] - m_new);
            m_run[r] = m_new;
            float rs = 0.0f;
#pragma unroll
            for (int t = 0; t < 4; ++t) {
                float p = __expf(st[t][r] - m_new);
                st[t][r] = p;
                rs += p;
            }
#pragma unroll
            for (int msk = 1; msk < 16; msk <<= 1)
                rs += __shfl_xor(rs, msk, 32);
            l_run[r] = l_run[r] * alpha[r] + rs;
        }
#pragma unroll
        for (int t = 0; t < 4; ++t)
#pragma unroll
            for (int r = 0; r < 8; ++r) o[t][r] *= alpha[r];

        // P: C-layout -> LDS (per-wave buffer; DS ops are wave-ordered)
#pragma unroll
        for (int t = 0; t < 4; ++t)
#pragma unroll
            for (int r = 0; r < 8; ++r)
                Pw[(size_t)(half * 8 + r) * 64 + t * 16 + nlane] =
                    f32_to_bf16(st[t][r]);

        // O += P @ V
#pragma unroll
        for (int kk = 0; kk < 2; ++kk) {
            v16bf pa = load_a_frag(Pw, 64, 0, kk * 32, lane);
#pragma unroll
            for (int t = 0; t < 4; ++t)
                o[t] = wmma_bf16(pa,
                                 load_b_frag_t(Vsb, 64, t * 16, kk * 32, lane),
                                 o[t]);
        }
        __syncthreads();            // buffer reads done before re-stage
    }

    // finalize: O /= l, write bf16 [s][d]
#pragma unroll
    for (int r = 0; r < 8; ++r) {
        float inv_l = 1.0f / l_run[r];
        int row = qs_w + half * 8 + r;
#pragma unroll
        for (int t = 0; t < 4; ++t) {
            int d = h * HDIM + t * 16 + nlane;
            Ob[(size_t)row * D_MODEL + d] = f32_to_bf16(o[t][r] * inv_l);
        }
    }
}

// ---- kernel 4: output projection + bias (fp32 out) -------------------------

__global__ void __launch_bounds__(256)
proj_gemm_kernel(const unsigned short* __restrict__ Ob,    // [4096][1024]
                 const unsigned short* __restrict__ wT,    // [1024][1024]
                 const float* __restrict__ bias,
                 float* __restrict__ out) {
    __shared__ unsigned short As[2 * 64 * 32];
    __shared__ unsigned short Bs[2 * 128 * 32];

    const int tid = threadIdx.x, lane = tid & 31, wave = tid >> 5;
    const int wm = wave >> 2, wn = wave & 3;
    const int bm = blockIdx.y, bn = blockIdx.x;

    GemmAcc g = gemm_mainloop(Ob, wT, As, Bs, bm, bn, tid, lane, wm, wn);

    const int half = lane >> 4, nlane = lane & 15;
    v8f acc[2][2] = {{g.c00, g.c01}, {g.c10, g.c11}};
#pragma unroll
    for (int ti = 0; ti < 2; ++ti)
#pragma unroll
        for (int tj = 0; tj < 2; ++tj)
#pragma unroll
            for (int r = 0; r < 8; ++r) {
                int row = bm * 64 + wm * 32 + ti * 16 + half * 8 + r;
                int col = bn * 128 + wn * 32 + tj * 16 + nlane;
                out[(size_t)row * D_MODEL + col] = acc[ti][tj][r] + bias[col];
            }
}

// ---- launcher --------------------------------------------------------------

extern "C" void kernel_launch(void* const* d_in, const int* in_sizes, int n_in,
                              void* d_out, int out_size, void* d_ws, size_t ws_size,
                              hipStream_t stream) {
    const float* x      = (const float*)d_in[0];   // [1,4096,1024]
    const float* w_qkv  = (const float*)d_in[1];   // [1024,3072]
    const float* b_qkv  = (const float*)d_in[2];   // [3072]
    const float* w_proj = (const float*)d_in[3];   // [1024,1024]
    const float* b_proj = (const float*)d_in[4];   // [1024]
    float* out = (float*)d_out;                    // [1,4096,1024]

    const size_t NX  = (size_t)S_LEN * D_MODEL;        // 4,194,304
    const size_t NWQ = (size_t)D_MODEL * 3 * D_MODEL;  // 3,145,728
    const size_t NWP = (size_t)D_MODEL * D_MODEL;      // 1,048,576

    unsigned short* p = (unsigned short*)d_ws;
    unsigned short* xb     = p; p += NX;   // [s][k] bf16
    unsigned short* wqkvT  = p; p += NWQ;  // [3072][1024] bf16 (transposed)
    unsigned short* wprojT = p; p += NWP;  // [1024][1024] bf16 (transposed)
    unsigned short* Qb     = p; p += NX;   // [h][s][hd] (pre-scaled)
    unsigned short* Kb     = p; p += NX;   // [h][s][hd]
    unsigned short* Vt     = p; p += NX;   // [h][hd][s]
    unsigned short* Ob     = p; p += NX;   // [s][d]

    cvt_f32_bf16_kernel<<<NX / 1024, 256, 0, stream>>>(x, xb, (int)NX);
    cvt_transpose_kernel<<<NWQ / 256, 256, 0, stream>>>(w_qkv, wqkvT, 3 * D_MODEL);
    cvt_transpose_kernel<<<NWP / 256, 256, 0, stream>>>(w_proj, wprojT, D_MODEL);

    qkv_gemm_kernel<<<dim3(3 * D_MODEL / 128, S_LEN / 64), 256, 0, stream>>>(
        xb, wqkvT, b_qkv, Qb, Kb, Vt);

    flash_attn_kernel<<<dim3(S_LEN / 128, NHEADS), 256, 0, stream>>>(Qb, Kb, Vt, Ob);

    proj_gemm_kernel<<<dim3(D_MODEL / 128, S_LEN / 64), 256, 0, stream>>>(
        Ob, wprojT, b_proj, out);
}
